// AnyStoryFluxAttnProcessor2_0_11330123727322
// MI455X (gfx1250) — matrix operand order repaired and checked by reading, not compiled, not run on hardware
//
#include <hip/hip_runtime.h>
#include <hip/hip_bf16.h>

// Problem constants (from the reference):
#define HH    24
#define DDIM  128
#define SEQ   3168
#define TEV   512      // txt end
#define TREV  576      // txt+redux end
#define TRIV  2624     // txt+redux+img end
#define TRIRV 3136     // +ref end
#define IMGV  2048

#define NELEM ((size_t)HH * SEQ * DDIM)      // 9,732,096 elements per tensor

typedef __bf16 bf16x16 __attribute__((ext_vector_type(16)));
typedef __bf16 bf16x8  __attribute__((ext_vector_type(8)));
typedef __bf16 bf16x4  __attribute__((ext_vector_type(4)));
typedef float  f32x8   __attribute__((ext_vector_type(8)));

static __device__ __forceinline__ f32x8 wmma_bf16(bf16x16 a, bf16x16 b, f32x8 c) {
    // D(16x16,f32) = A(16x32,bf16) x B(32x16,bf16) + C
    return __builtin_amdgcn_wmma_f32_16x16x32_bf16(false, a, false, b, (short)0, c,
                                                   false, false);
}
static __device__ __forceinline__ bf16x16 cat8(bf16x8 a, bf16x8 b) {
    return __builtin_shufflevector(a, b, 0,1,2,3,4,5,6,7,8,9,10,11,12,13,14,15);
}
static __device__ __forceinline__ bf16x16 ld32(const __bf16* p) {  // 32B, 16B-aligned
    return cat8(*(const bf16x8*)p, *(const bf16x8*)(p + 8));
}

// ---------------- pre-pass 1: elementwise fp32 -> bf16 (Q pre-scaled, K) ----------
__global__ __launch_bounds__(256)
void cvt_qk_kernel(const float* __restrict__ q, const float* __restrict__ k,
                   __bf16* __restrict__ qw, __bf16* __restrict__ kw) {
    const float scale = 0.08838834764831845f;    // 1/sqrt(128)
    size_t i = ((size_t)blockIdx.x * 256 + threadIdx.x) * 4;
    float4 a = *(const float4*)(q + i);
    float4 b = *(const float4*)(k + i);
    bf16x4 qa = { (__bf16)(a.x * scale), (__bf16)(a.y * scale),
                  (__bf16)(a.z * scale), (__bf16)(a.w * scale) };
    bf16x4 kb = { (__bf16)b.x, (__bf16)b.y, (__bf16)b.z, (__bf16)b.w };
    *(bf16x4*)(qw + i) = qa;
    *(bf16x4*)(kw + i) = kb;
}

// ---------------- pre-pass 2: V fp32 [h][s][d] -> bf16 transposed [h][d][s] -------
__global__ __launch_bounds__(256)
void cvt_v_transpose_kernel(const float* __restrict__ v, __bf16* __restrict__ vw) {
    __shared__ __bf16 tile[DDIM * 33];           // 32(s) x 128(d) tile, padded
    const int tid = threadIdx.x;
    const int h  = blockIdx.x / (SEQ / 32);
    const int s0 = (blockIdx.x % (SEQ / 32)) * 32;

    #pragma unroll
    for (int it = 0; it < 16; it++) {            // coalesced read over d
        int e = it * 256 + tid;
        int sl = e >> 7, d = e & 127;
        tile[d * 33 + sl] = (__bf16)v[((size_t)h * SEQ + s0 + sl) * DDIM + d];
    }
    __syncthreads();
    #pragma unroll
    for (int it = 0; it < 16; it++) {            // coalesced write over s
        int e = it * 256 + tid;
        int d = e >> 5, sl = e & 31;
        vw[((size_t)h * DDIM + d) * SEQ + s0 + sl] = tile[d * 33 + sl];
    }
}

// ---------------- fast path: flash attention, Bc=64, all-bf16 operands ------------
#define PPAD 72   // LDS stride for 16 x 64 P staging (multiple of 8 for 16B aligns)

__global__ __launch_bounds__(32)
void flux_attn_fast(const __bf16* __restrict__ qw, const __bf16* __restrict__ kw,
                    const __bf16* __restrict__ vw, const float* __restrict__ refm,
                    const float* __restrict__ routing, float* __restrict__ out)
{
    __shared__ __bf16 pl[16 * PPAD];

    const int lane = threadIdx.x;
    const int half = lane >> 4;
    const int lh   = lane & 15;

    int tile = blockIdx.x;
    int prob, h, qbase, keybase = 0, nsteps;
    if (tile < HH * 164) {                         // out1
        prob = 0; h = tile / 164; qbase = (tile % 164) * 16; nsteps = TRIRV / 64;   // 49
    } else if (tile < HH * 164 + HH * 32) {        // ref_out (2 groups of 256)
        int t = tile - HH * 164; prob = 1; h = t / 32;
        int lt = t % 32; int g = lt >> 4;
        qbase = TRIV + g * 256 + (lt & 15) * 16;
        keybase = TRIV + g * 256; nsteps = 4;
    } else {                                       // router_out (img 2048 + router 32)
        int t = tile - HH * 164 - HH * 32; prob = 2; h = t >> 1;
        qbase = TRIRV + (t & 1) * 16; nsteps = 33; // 32 img steps + 1 tail step
    }

    // Q A-fragments: resident, loaded directly as bf16
    bf16x16 qa[4];
    {
        const __bf16* qp = qw + ((size_t)h * SEQ + qbase + lh) * DDIM;
        #pragma unroll
        for (int c = 0; c < 4; c++)
            qa[c] = cat8(*(const bf16x8*)(qp + c * 32 + 16 * half),
                         *(const bf16x8*)(qp + c * 32 + 16 * half + 8));
    }

    // routing mask terms per owned row (rows m = r + 8*half), prob 0 only
    float rt0[8], rt1[8];
    bool  qimg[8];
    #pragma unroll
    for (int r = 0; r < 8; r++) {
        int qr = qbase + r + 8 * half;
        bool img = (prob == 0) && (qr >= TREV) && (qr < TRIV);
        qimg[r] = img;
        rt0[r] = img ? (routing[qr - TREV] - 1.0f) * 100.0f : 0.0f;
        rt1[r] = img ? (routing[IMGV + qr - TREV] - 1.0f) * 100.0f : 0.0f;
    }

    f32x8 O[8];
    #pragma unroll
    for (int g = 0; g < 8; g++) O[g] = (f32x8){0,0,0,0,0,0,0,0};
    float mrow[8], lpart[8];                       // lpart: per-LANE partial row sums
    #pragma unroll
    for (int r = 0; r < 8; r++) { mrow[r] = -1e30f; lpart[r] = 0.0f; }

    for (int s = 0; s < nsteps; s++) {
        int k0;
        if      (prob == 0) k0 = s * 64;
        else if (prob == 1) k0 = keybase + s * 64;
        else                k0 = (s < 32) ? (TREV + s * 64) : (TRIRV - 32);

        // ---- S = Q K^T : 4 key tiles x 4 d-chunks ----
        f32x8 St[4];
        #pragma unroll
        for (int t = 0; t < 4; t++) {
            f32x8 acc = (f32x8){0,0,0,0,0,0,0,0};
            const __bf16* kp = kw + ((size_t)h * SEQ + k0 + t * 16 + lh) * DDIM + 16 * half;
            #pragma unroll
            for (int c = 0; c < 4; c++)
                acc = wmma_bf16(qa[c], ld32(kp + c * 32), acc);
            St[t] = acc;
        }

        // ---- additive mask (prob 0; all region boundaries are 64-aligned) ----
        if (prob == 0) {
            #pragma unroll
            for (int t = 0; t < 4; t++) {
                int kk = k0 + t * 16 + lh;
                if (kk >= TRIV) {
                    int kr = kk - TRIV;
                    const float* rmp = refm + (size_t)kr * TRIV + qbase + 8 * half;
                    bool c1 = (kr >> 8) != 0;
                    #pragma unroll
                    for (int r = 0; r < 8; r++) {
                        float add = qimg[r] ? (c1 ? rt1[r] : rt0[r]) : 0.0f;
                        St[t][r] += 1.5f + (rmp[r] - 1.0f) * 100.0f + add;
                    }
                } else if (kk >= TEV && kk < TREV) {
                    bool c1 = ((kk - TEV) >> 5) != 0;
                    #pragma unroll
                    for (int r = 0; r < 8; r++)
                        if (qimg[r]) St[t][r] += (c1 ? rt1[r] : rt0[r]);
                }
            }
        }
        // router tail step: tiles 0,1 are re-read ref keys -> mask out (contribute 0)
        if (prob == 2 && s == 32) {
            #pragma unroll
            for (int r = 0; r < 8; r++) { St[0][r] = -1e30f; St[1][r] = -1e30f; }
        }

        // ---- row max (only reduction per step; sums deferred to epilogue) ----
        float mx[8];
        bool upd = false;
        #pragma unroll
        for (int r = 0; r < 8; r++) {
            float m01 = fmaxf(St[0][r], St[1][r]);
            float m23 = fmaxf(St[2][r], St[3][r]);
            float m = fmaxf(m01, m23);
            #pragma unroll
            for (int msk = 1; msk < 16; msk <<= 1) m = fmaxf(m, __shfl_xor(m, msk, 32));
            mx[r] = m;
            upd = upd || (m > mrow[r]);
        }
        if (__any(upd)) {                          // uniform branch; rare after warmup
            float al[8];
            #pragma unroll
            for (int r = 0; r < 8; r++) {
                float mn = fmaxf(mrow[r], mx[r]);
                al[r] = __expf(mrow[r] - mn);
                mrow[r] = mn;
                lpart[r] *= al[r];
            }
            #pragma unroll
            for (int g = 0; g < 8; g++)
                #pragma unroll
                for (int r = 0; r < 8; r++) O[g][r] *= al[r];
        }

        // ---- P = exp(S - m), per-lane partial sums, stage P to LDS row-major ----
        #pragma unroll
        for (int r = 0; r < 8; r++) {
            int m = r + 8 * half;
            float e0 = __expf(St[0][r] - mrow[r]);
            float e1 = __expf(St[1][r] - mrow[r]);
            float e2 = __expf(St[2][r] - mrow[r]);
            float e3 = __expf(St[3][r] - mrow[r]);
            lpart[r] += (e0 + e1) + (e2 + e3);
            pl[m * PPAD + lh]      = (__bf16)e0;
            pl[m * PPAD + 16 + lh] = (__bf16)e1;
            pl[m * PPAD + 32 + lh] = (__bf16)e2;
            pl[m * PPAD + 48 + lh] = (__bf16)e3;
        }
        asm volatile("s_wait_dscnt 0x0" ::: "memory");

        // P A-fragments (k 0..31 and 32..63)
        const __bf16* pr = &pl[lh * PPAD + 16 * half];
        bf16x16 pa0 = cat8(*(const bf16x8*)(pr),      *(const bf16x8*)(pr + 8));
        bf16x16 pa1 = cat8(*(const bf16x8*)(pr + 32), *(const bf16x8*)(pr + 40));

        // ---- O += P x V : V B-fragments direct from pre-transposed bf16 global ----
        #pragma unroll
        for (int g = 0; g < 8; g++) {
            const __bf16* vp = vw + ((size_t)h * DDIM + g * 16 + lh) * SEQ + k0 + 16 * half;
            O[g] = wmma_bf16(pa0, ld32(vp),      O[g]);
            O[g] = wmma_bf16(pa1, ld32(vp + 32), O[g]);
        }
        asm volatile("s_wait_dscnt 0x0" ::: "memory");  // pl reused next iteration
    }

    // ---- epilogue: cross-lane row-sum reduction (once), normalize, store ----
    #pragma unroll
    for (int r = 0; r < 8; r++) {
        float l = lpart[r];
        #pragma unroll
        for (int msk = 1; msk < 16; msk <<= 1) l += __shfl_xor(l, msk, 32);
        float inv = 1.0f / l;
        float* op = out + ((size_t)h * SEQ + qbase + r + 8 * half) * DDIM + lh;
        #pragma unroll
        for (int g = 0; g < 8; g++) op[g * 16] = O[g][r] * inv;
    }
}

// ---------------- fallback (self-contained fp32->bf16 in-kernel, Bc=32) -----------
#define KPAD 40
#define FPPAD 40

__global__ __launch_bounds__(32)
void flux_attn_fallback(const float* __restrict__ qg, const float* __restrict__ kg,
                        const float* __restrict__ vg, const float* __restrict__ refm,
                        const float* __restrict__ routing, float* __restrict__ out)
{
    __shared__ __bf16 vt[DDIM * KPAD];
    __shared__ __bf16 pl[16 * FPPAD];

    const int lane = threadIdx.x;
    const int half = lane >> 4;
    const int lh   = lane & 15;

    int tile = blockIdx.x;
    int prob, h, qbase, keybase = 0, nsteps;
    if (tile < HH * 164) {
        prob = 0; h = tile / 164; qbase = (tile % 164) * 16; nsteps = TRIRV / 32;
    } else if (tile < HH * 164 + HH * 32) {
        int t = tile - HH * 164; prob = 1; h = t / 32;
        int lt = t % 32; int g = lt >> 4;
        qbase = TRIV + g * 256 + (lt & 15) * 16;
        keybase = TRIV + g * 256; nsteps = 8;
    } else {
        int t = tile - HH * 164 - HH * 32; prob = 2; h = t >> 1;
        qbase = TRIRV + (t & 1) * 16; nsteps = 65;
    }

    const float scale = 0.08838834764831845f;

    bf16x16 qa[4];
    {
        const float* qp = qg + ((size_t)h * SEQ + qbase + lh) * DDIM;
        #pragma unroll
        for (int c = 0; c < 4; c++) {
            const float* p0 = qp + c * 32 + 16 * half;
            const float* p1 = qp + c * 32 + 16 * half + 8;
            #pragma unroll
            for (int i = 0; i < 8; i++) {
                qa[c][i]     = (__bf16)(p0[i] * scale);
                qa[c][8 + i] = (__bf16)(p1[i] * scale);
            }
        }
    }

    float rt0[8], rt1[8];
    bool  qimg[8];
    #pragma unroll
    for (int r = 0; r < 8; r++) {
        int qr = qbase + r + 8 * half;
        bool img = (prob == 0) && (qr >= TREV) && (qr < TRIV);
        qimg[r] = img;
        rt0[r] = img ? (routing[qr - TREV] - 1.0f) * 100.0f : 0.0f;
        rt1[r] = img ? (routing[IMGV + qr - TREV] - 1.0f) * 100.0f : 0.0f;
    }

    f32x8 O[8];
    #pragma unroll
    for (int g = 0; g < 8; g++) O[g] = (f32x8){0,0,0,0,0,0,0,0};
    float mrow[8], lrow[8];
    #pragma unroll
    for (int r = 0; r < 8; r++) { mrow[r] = -1e30f; lrow[r] = 0.0f; }

    for (int s = 0; s < nsteps; s++) {
        int k0;
        if      (prob == 0) k0 = s * 32;
        else if (prob == 1) k0 = keybase + s * 32;
        else                k0 = (s < 64) ? (TREV + s * 32) : TRIRV;

        {
            const float* vp = vg + ((size_t)h * SEQ + k0 + lane) * DDIM;
            #pragma unroll
            for (int d = 0; d < DDIM; d += 4) {
                float4 val = *(const float4*)(vp + d);
                vt[(d + 0) * KPAD + lane] = (__bf16)val.x;
                vt[(d + 1) * KPAD + lane] = (__bf16)val.y;
                vt[(d + 2) * KPAD + lane] = (__bf16)val.z;
                vt[(d + 3) * KPAD + lane] = (__bf16)val.w;
            }
        }

        f32x8 St[2];
        #pragma unroll
        for (int t = 0; t < 2; t++) {
            f32x8 acc = (f32x8){0,0,0,0,0,0,0,0};
            const float* kp = kg + ((size_t)h * SEQ + k0 + t * 16 + lh) * DDIM + 16 * half;
            #pragma unroll
            for (int c = 0; c < 4; c++) {
                bf16x16 kb;
                const float* p = kp + c * 32;
                #pragma unroll
                for (int i = 0; i < 16; i++) kb[i] = (__bf16)p[i];
                acc = wmma_bf16(qa[c], kb, acc);
            }
            St[t] = acc;
        }

        if (prob == 0) {
            #pragma unroll
            for (int t = 0; t < 2; t++) {
                int kk = k0 + t * 16 + lh;
                if (kk >= TRIV) {
                    int kr = kk - TRIV;
                    const float* rmp = refm + (size_t)kr * TRIV + qbase + 8 * half;
                    bool c1 = (kr >> 8) != 0;
                    #pragma unroll
                    for (int r = 0; r < 8; r++) {
                        float add = qimg[r] ? (c1 ? rt1[r] : rt0[r]) : 0.0f;
                        St[t][r] += 1.5f + (rmp[r] - 1.0f) * 100.0f + add;
                    }
                } else if (kk >= TEV && kk < TREV) {
                    bool c1 = ((kk - TEV) >> 5) != 0;
                    #pragma unroll
                    for (int r = 0; r < 8; r++)
                        if (qimg[r]) St[t][r] += (c1 ? rt1[r] : rt0[r]);
                }
            }
        }

        float p0v[8], p1v[8], alpha[8];
        #pragma unroll
        for (int r = 0; r < 8; r++) {
            float mxv = fmaxf(St[0][r], St[1][r]);
            #pragma unroll
            for (int msk = 1; msk < 16; msk <<= 1) mxv = fmaxf(mxv, __shfl_xor(mxv, msk, 32));
            float mn = fmaxf(mrow[r], mxv);
            alpha[r] = __expf(mrow[r] - mn);
            float e0 = __expf(St[0][r] - mn);
            float e1 = __expf(St[1][r] - mn);
            float rs = e0 + e1;
            #pragma unroll
            for (int msk = 1; msk < 16; msk <<= 1) rs += __shfl_xor(rs, msk, 32);
            lrow[r] = lrow[r] * alpha[r] + rs;
            mrow[r] = mn;
            p0v[r] = e0; p1v[r] = e1;
        }
        #pragma unroll
        for (int g = 0; g < 8; g++)
            #pragma unroll
            for (int r = 0; r < 8; r++) O[g][r] *= alpha[r];

        #pragma unroll
        for (int r = 0; r < 8; r++) {
            int m = r + 8 * half;
            pl[m * FPPAD + lh]      = (__bf16)p0v[r];
            pl[m * FPPAD + 16 + lh] = (__bf16)p1v[r];
        }
        asm volatile("s_wait_dscnt 0x0" ::: "memory");

        bf16x16 pa = cat8(*(const bf16x8*)&pl[lh * FPPAD + 16 * half],
                          *(const bf16x8*)&pl[lh * FPPAD + 16 * half + 8]);

        #pragma unroll
        for (int g = 0; g < 8; g++) {
            const __bf16* vp = &vt[(g * 16 + lh) * KPAD + 16 * half];
            O[g] = wmma_bf16(pa, ld32(vp), O[g]);
        }
        asm volatile("s_wait_dscnt 0x0" ::: "memory");
    }

    #pragma unroll
    for (int r = 0; r < 8; r++) {
        float inv = 1.0f / lrow[r];
        float* op = out + ((size_t)h * SEQ + qbase + r + 8 * half) * DDIM + lh;
        #pragma unroll
        for (int g = 0; g < 8; g++) op[g * 16] = O[g][r] * inv;
    }
}

// -----------------------------------------------------------------------------------
extern "C" void kernel_launch(void* const* d_in, const int* in_sizes, int n_in,
                              void* d_out, int out_size, void* d_ws, size_t ws_size,
                              hipStream_t stream) {
    (void)in_sizes; (void)n_in; (void)out_size;
    const float* q  = (const float*)d_in[0];
    const float* k  = (const float*)d_in[1];
    const float* v  = (const float*)d_in[2];
    const float* rm = (const float*)d_in[3];
    const float* rt = (const float*)d_in[4];
    float* out = (float*)d_out;

    const int ntiles = HH * 164 + HH * 32 + HH * 2;       // 4752 query tiles
    const size_t ws_needed = 3 * NELEM * sizeof(__bf16);  // ~58.4 MB

    if (ws_size >= ws_needed) {
        __bf16* qw = (__bf16*)d_ws;
        __bf16* kw = qw + NELEM;
        __bf16* vw = kw + NELEM;
        // pre-pass: bf16 copies (Q pre-scaled; V transposed per head)
        cvt_qk_kernel<<<(int)(NELEM / (256 * 4)), 256, 0, stream>>>(q, k, qw, kw);
        cvt_v_transpose_kernel<<<HH * (SEQ / 32), 256, 0, stream>>>(v, vw);
        flux_attn_fast<<<ntiles, 32, 0, stream>>>(qw, kw, vw, rm, rt, out);
    } else {
        flux_attn_fallback<<<ntiles, 32, 0, stream>>>(q, k, v, rm, rt, out);
    }
}
